// SEDD_flex_88416196755850
// MI455X (gfx1250) — matrix operand order repaired and checked by reading, not compiled, not run on hardware
//
#include <hip/hip_runtime.h>
#include <cstdint>
#include <cstddef>

// ---------------------------------------------------------------------------
// SEDD transformer forward for MI455X (gfx1250, wave32, WMMA).
// All large GEMMs use v_wmma_f32_16x16x32_bf16 with fp32 accumulation,
// register-blocked 32x32 per wave (4 WMMAs per K-chunk, 2 A + 2 B frags).
// ---------------------------------------------------------------------------

#define TT   3072      // total tokens
#define DD   768       // model dim
#define HH   12        // heads
#define HDD  64        // head dim
#define VV   32001     // vocab
#define VPAD 32128     // vocab padded to 128
#define CC   128       // cond dim
#define BB   8         // batch (num sequences)

typedef __attribute__((ext_vector_type(8)))  float        v8f;
typedef __attribute__((ext_vector_type(8)))  unsigned int v8u;
typedef __attribute__((ext_vector_type(16))) __bf16       v16bf;

__constant__ int d_seq_start[8] = {0, 512, 896, 1152, 1792, 2240, 2560, 2752};
__constant__ int d_seq_len[8]   = {512, 384, 256, 640, 448, 320, 192, 320};

__device__ __forceinline__ unsigned short f2bf(float f) {
    unsigned int u = __builtin_bit_cast(unsigned int, f);
    unsigned int r = u + 0x7fffu + ((u >> 16) & 1u);   // round-to-nearest-even
    return (unsigned short)(r >> 16);
}

__device__ __forceinline__ float gelu_tanh(float x) {
    float x3 = x * x * x;
    return 0.5f * x * (1.f + tanhf(0.79788456080286536f * (x + 0.044715f * x3)));
}

__device__ __forceinline__ float silu(float x) {
    return x / (1.f + __expf(-x));
}

// ---------------------------------------------------------------------------
// Embedding gather: x[t, d] = emb[indices[t], d]
// ---------------------------------------------------------------------------
__global__ void embed_kernel(const int* __restrict__ indices,
                             const float* __restrict__ emb,
                             float* __restrict__ x) {
    int idx = blockIdx.x * blockDim.x + threadIdx.x;
    if (idx >= TT * DD) return;
    int t = idx / DD, d = idx % DD;
    x[idx] = emb[(size_t)indices[t] * DD + d];
}

// ---------------------------------------------------------------------------
// Timestep conditioning: c = silu(silu(temb @ w1 + b1) @ w2 + b2)
// rowsub[b] = log(expm1(sigma)) + log(V-1)
// One block of 128 threads per sequence.
// ---------------------------------------------------------------------------
__global__ void timestep_kernel(const float* __restrict__ sigma,
                                const float* __restrict__ w1, const float* __restrict__ b1,
                                const float* __restrict__ w2, const float* __restrict__ b2,
                                float* __restrict__ c, float* __restrict__ rowsub) {
    int b = blockIdx.x;
    int tid = threadIdx.x;
    __shared__ float temb[256];
    __shared__ float h1[128];
    float sg = sigma[b];
    float fr = __expf(-logf(10000.f) * (float)tid / 128.f);
    float a = sg * fr;
    temb[tid]       = cosf(a);
    temb[tid + 128] = sinf(a);
    __syncthreads();
    float acc = b1[tid];
    for (int k = 0; k < 256; ++k) acc += temb[k] * w1[k * CC + tid];
    h1[tid] = silu(acc);
    __syncthreads();
    float acc2 = b2[tid];
    for (int k = 0; k < 128; ++k) acc2 += h1[k] * w2[k * CC + tid];
    c[b * CC + tid] = silu(acc2);
    if (tid == 0) rowsub[b] = logf(expm1f(sg)) + logf((float)(VV - 1));
}

// ---------------------------------------------------------------------------
// Small dense: out[b, j] = sum_k c[b, k] * W[k, j] + bias[j]   (K = 128)
// ---------------------------------------------------------------------------
__global__ void small_mm_kernel(const float* __restrict__ c,
                                const float* __restrict__ W,
                                const float* __restrict__ bias,
                                float* __restrict__ out, int N) {
    int idx = blockIdx.x * blockDim.x + threadIdx.x;
    if (idx >= BB * N) return;
    int b = idx / N, j = idx % N;
    float acc = bias[j];
    const float* cb = c + b * CC;
    for (int k = 0; k < CC; ++k) acc += cb[k] * W[(size_t)k * N + j];
    out[idx] = acc;
}

// ---------------------------------------------------------------------------
// LayerNorm + adaLN modulation, bf16 output.
// h = LN(x)*w+b; out = h*(1 + ada[seq][scOff+i]) + ada[seq][shOff+i]
// One block (256 thr) per token; D = 768 = 3*256.
// ---------------------------------------------------------------------------
__global__ __launch_bounds__(256) void ln_mod_kernel(
        const float* __restrict__ x,
        const float* __restrict__ w, const float* __restrict__ b,
        const float* __restrict__ ada, int adaStride, int shOff, int scOff,
        const int* __restrict__ seq_idx,
        unsigned short* __restrict__ out) {
    int t = blockIdx.x;
    int tid = threadIdx.x;
    const float* xt = x + (size_t)t * DD;
    __shared__ float r1[256], r2[256];
    float s = 0.f, s2 = 0.f;
    for (int i = tid; i < DD; i += 256) { float v = xt[i]; s += v; s2 += v * v; }
    r1[tid] = s; r2[tid] = s2;
    __syncthreads();
    for (int off = 128; off > 0; off >>= 1) {
        if (tid < off) { r1[tid] += r1[tid + off]; r2[tid] += r2[tid + off]; }
        __syncthreads();
    }
    float mu = r1[0] * (1.f / DD);
    float var = r2[0] * (1.f / DD) - mu * mu;
    float rstd = rsqrtf(var + 1e-5f);
    const float* ar = ada + (size_t)seq_idx[t] * adaStride;
    unsigned short* ot = out + (size_t)t * DD;
    for (int i = tid; i < DD; i += 256) {
        float v = (xt[i] - mu) * rstd * w[i] + b[i];
        float h = v * (1.f + ar[scOff + i]) + ar[shOff + i];
        ot[i] = f2bf(h);
    }
}

// ---------------------------------------------------------------------------
// fp32 -> bf16 (row-major, no transpose)
// ---------------------------------------------------------------------------
__global__ void f32_to_bf16_kernel(const float* __restrict__ src,
                                   unsigned short* __restrict__ dst, int n) {
    int i = blockIdx.x * blockDim.x + threadIdx.x;
    if (i < n) dst[i] = f2bf(src[i]);
}

// ---------------------------------------------------------------------------
// Weight convert + transpose: W[K,N] fp32 (row-major) -> Bt[Npad,K] bf16,
// zero-padded rows for n >= N.
// ---------------------------------------------------------------------------
__global__ void conv_tr_kernel(const float* __restrict__ W,
                               unsigned short* __restrict__ Bt,
                               int K, int N, int Npad) {
    int idx = blockIdx.x * blockDim.x + threadIdx.x;
    if (idx >= Npad * K) return;
    int n = idx / K, k = idx % K;
    Bt[idx] = (n < N) ? f2bf(W[(size_t)k * N + n]) : (unsigned short)0;
}

// ---------------------------------------------------------------------------
// Fragment loaders (CDNA5 ISA 7.12.2 layouts, bf16 packed as dwords).
// A 16x32 (MxK): lane l<16 row M=l, K pairs 0,2,4,6 / 16,18,20,22 (+half*8)
// B 32x16 (KxN): lane l<16 col N=l, K pairs 0,2,4,6 / 16,18,20,22 (+half*8)
// ---------------------------------------------------------------------------
__device__ __forceinline__ v16bf load_a_frag(const unsigned short* Arow, int kb, int half) {
    int ka = kb + half * 8;
    v8u p;
    p[0] = *(const unsigned int*)(Arow + ka + 0);
    p[1] = *(const unsigned int*)(Arow + ka + 2);
    p[2] = *(const unsigned int*)(Arow + ka + 4);
    p[3] = *(const unsigned int*)(Arow + ka + 6);
    p[4] = *(const unsigned int*)(Arow + ka + 16);
    p[5] = *(const unsigned int*)(Arow + ka + 18);
    p[6] = *(const unsigned int*)(Arow + ka + 20);
    p[7] = *(const unsigned int*)(Arow + ka + 22);
    return __builtin_bit_cast(v16bf, p);
}

__device__ __forceinline__ v16bf load_b_frag(const unsigned short* Bcol, int kb, int half) {
    int kB = kb + half * 8;
    v8u p;
    p[0] = *(const unsigned int*)(Bcol + kB + 0);
    p[1] = *(const unsigned int*)(Bcol + kB + 2);
    p[2] = *(const unsigned int*)(Bcol + kB + 4);
    p[3] = *(const unsigned int*)(Bcol + kB + 6);
    p[4] = *(const unsigned int*)(Bcol + kB + 16);
    p[5] = *(const unsigned int*)(Bcol + kB + 18);
    p[6] = *(const unsigned int*)(Bcol + kB + 20);
    p[7] = *(const unsigned int*)(Bcol + kB + 22);
    return __builtin_bit_cast(v16bf, p);
}

// ---------------------------------------------------------------------------
// WMMA bf16 GEMM: Out[M,N] = A[M,K] @ B[K,N] (+ epilogue).
// A: bf16 row-major, Bt: bf16 col-major (padded to grid N).
// Block = 256 threads = 8 waves arranged 2(M) x 4(N); each wave computes a
// 32x32 output tile (2x2 accumulators), so block tile = 64(M) x 128(N).
// Per K-chunk: 2 A-frags + 2 B-frags feed 4 WMMAs (2x data reuse vs 1x1).
// MODE 0: out = acc (+bias)
// MODE 1: out = gelu(acc + bias)
// MODE 2: out = gate[seq[m]][gOff+n] * (acc (+bias)) + resid[m][n]
// MODE 3: out = acc + bias[n] - rowsub[seq[m]]
// ---------------------------------------------------------------------------
template <int MODE>
__global__ __launch_bounds__(256) void gemm_bf16_wmma(
        const unsigned short* __restrict__ A,
        const unsigned short* __restrict__ Bt,
        const float* __restrict__ bias,
        const int* __restrict__ seq,
        const float* __restrict__ gate, int gOff, int gStride,
        const float* __restrict__ resid,
        const float* __restrict__ rowsub,
        float* __restrict__ Out,
        int N, int K) {
    int tid  = threadIdx.x;
    int wave = tid >> 5;
    int lane = tid & 31;
    int Mbase = blockIdx.y * 64  + (wave >> 2) * 32;   // 2 waves in M
    int Nbase = blockIdx.x * 128 + (wave & 3) * 32;    // 4 waves in N
    int half = lane >> 4;          // 0: lanes 0-15, 1: lanes 16-31
    int l16  = lane & 15;

    const unsigned short* Arow0 = A  + (size_t)(Mbase + l16) * K;
    const unsigned short* Arow1 = Arow0 + (size_t)16 * K;
    const unsigned short* Bcol0 = Bt + (size_t)(Nbase + l16) * K;
    const unsigned short* Bcol1 = Bcol0 + (size_t)16 * K;

    v8f acc00 = {0.f, 0.f, 0.f, 0.f, 0.f, 0.f, 0.f, 0.f};
    v8f acc01 = acc00, acc10 = acc00, acc11 = acc00;

    for (int kb = 0; kb < K; kb += 32) {
        v16bf a0 = load_a_frag(Arow0, kb, half);
        v16bf a1 = load_a_frag(Arow1, kb, half);
        v16bf b0 = load_b_frag(Bcol0, kb, half);
        v16bf b1 = load_b_frag(Bcol1, kb, half);
        acc00 = __builtin_amdgcn_wmma_f32_16x16x32_bf16(false, a0, false, b0, (short)0, acc00, false, false);
        acc01 = __builtin_amdgcn_wmma_f32_16x16x32_bf16(false, a0, false, b1, (short)0, acc01, false, false);
        acc10 = __builtin_amdgcn_wmma_f32_16x16x32_bf16(false, a1, false, b0, (short)0, acc10, false, false);
        acc11 = __builtin_amdgcn_wmma_f32_16x16x32_bf16(false, a1, false, b1, (short)0, acc11, false, false);
    }

    // D layout (f32 16x16): lane<16 -> (M=r, N=lane); lane>=16 -> (M=r+8, N=lane-16)
#pragma unroll
    for (int mi = 0; mi < 2; ++mi) {
#pragma unroll
        for (int ni = 0; ni < 2; ++ni) {
            const v8f& acc = (mi == 0) ? (ni == 0 ? acc00 : acc01)
                                       : (ni == 0 ? acc10 : acc11);
            int n = Nbase + ni * 16 + l16;
            if (n >= N) continue;
#pragma unroll
            for (int r = 0; r < 8; ++r) {
                int m = Mbase + mi * 16 + half * 8 + r;
                float v = acc[r];
                if (MODE == 0) {
                    if (bias) v += bias[n];
                    Out[(size_t)m * N + n] = v;
                } else if (MODE == 1) {
                    v += bias[n];
                    Out[(size_t)m * N + n] = gelu_tanh(v);
                } else if (MODE == 2) {
                    if (bias) v += bias[n];
                    float g = gate[(size_t)seq[m] * gStride + gOff + n];
                    Out[(size_t)m * N + n] = g * v + resid[(size_t)m * N + n];
                } else { // MODE == 3
                    v += bias[n] - rowsub[seq[m]];
                    Out[(size_t)m * N + n] = v;
                }
            }
        }
    }
}

// ---------------------------------------------------------------------------
// RoPE applied in-place to q and k inside qkv[T, 3*D].
// One thread per (t, j in {q,k}, head, i < 32).
// ---------------------------------------------------------------------------
__global__ void rope_kernel(float* __restrict__ qkv,
                            const int* __restrict__ positions) {
    int idx = blockIdx.x * blockDim.x + threadIdx.x;
    if (idx >= TT * 2 * HH * 32) return;
    int i = idx & 31;
    int tmp = idx >> 5;
    int h = tmp % HH; tmp /= HH;
    int j = tmp & 1;
    int t = tmp >> 1;
    float p = (float)positions[t];
    float inv = __expf(-logf(10000.f) * (float)i / 32.f);  // 10000^{-i/32}
    float a = p * inv;
    float cs = cosf(a), sn = sinf(a);
    size_t base = (size_t)t * (3 * DD) + (size_t)j * DD + h * HDD;
    float x1 = qkv[base + i];
    float x2 = qkv[base + 32 + i];
    qkv[base + i]      = x1 * cs - x2 * sn;
    qkv[base + 32 + i] = x2 * cs + x1 * sn;
}

// ---------------------------------------------------------------------------
// Block-diagonal attention (one block per (head, sequence)); softmax in LDS.
// qkv layout: [T, 3*D], q at off 0, k at off D, v at off 2D, head h at h*64.
// Writes o[T, D] (thd flattened).
// ---------------------------------------------------------------------------
__global__ __launch_bounds__(256) void attn_kernel(const float* __restrict__ qkv,
                                                   float* __restrict__ o) {
    int h = blockIdx.x;
    int b = blockIdx.y;
    int start = d_seq_start[b];
    int len   = d_seq_len[b];
    int tid = threadIdx.x;
    __shared__ float sc[640];
    __shared__ float qv[64];
    __shared__ float red[256];

    for (int q = 0; q < len; ++q) {
        int tq = start + q;
        if (tid < 64) qv[tid] = qkv[(size_t)tq * (3 * DD) + h * HDD + tid];
        __syncthreads();
        // scores + local max
        float lmax = -1e30f;
        for (int j = tid; j < len; j += 256) {
            const float* kp = qkv + (size_t)(start + j) * (3 * DD) + DD + h * HDD;
            float s = 0.f;
#pragma unroll 8
            for (int d = 0; d < HDD; ++d) s += qv[d] * kp[d];
            s *= 0.125f;  // 1/sqrt(64)
            sc[j] = s;
            lmax = fmaxf(lmax, s);
        }
        red[tid] = lmax;
        __syncthreads();
        for (int off = 128; off > 0; off >>= 1) {
            if (tid < off) red[tid] = fmaxf(red[tid], red[tid + off]);
            __syncthreads();
        }
        float mx = red[0];
        __syncthreads();
        // exp + local sum
        float lsum = 0.f;
        for (int j = tid; j < len; j += 256) {
            float e = __expf(sc[j] - mx);
            sc[j] = e;
            lsum += e;
        }
        red[tid] = lsum;
        __syncthreads();
        for (int off = 128; off > 0; off >>= 1) {
            if (tid < off) red[tid] += red[tid + off];
            __syncthreads();
        }
        float invs = 1.f / red[0];
        __syncthreads();
        // weighted value sum: 4 partials per output dim
        int d = tid & 63;
        int part = tid >> 6;
        float acc = 0.f;
        for (int j = part; j < len; j += 4)
            acc += sc[j] * qkv[(size_t)(start + j) * (3 * DD) + 2 * DD + h * HDD + d];
        red[tid] = acc;
        __syncthreads();
        if (part == 0)
            o[(size_t)tq * DD + h * HDD + d] =
                (red[d] + red[64 + d] + red[128 + d] + red[192 + d]) * invs;
        __syncthreads();
    }
}

// ---------------------------------------------------------------------------
// out[t, indices[t]] = 0
// ---------------------------------------------------------------------------
__global__ void set_targets_kernel(float* __restrict__ out,
                                   const int* __restrict__ indices) {
    int t = blockIdx.x * blockDim.x + threadIdx.x;
    if (t < TT) out[(size_t)t * VV + indices[t]] = 0.f;
}

// ---------------------------------------------------------------------------
// Host orchestration
// ---------------------------------------------------------------------------
static inline int cdiv(int a, int b) { return (a + b - 1) / b; }

extern "C" void kernel_launch(void* const* d_in, const int* in_sizes, int n_in,
                              void* d_out, int out_size, void* d_ws, size_t ws_size,
                              hipStream_t stream) {
    (void)in_sizes; (void)n_in; (void)out_size; (void)ws_size;
    const int*   indices   = (const int*)  d_in[0];
    const float* sigma     = (const float*)d_in[1];
    const int*   seq_idx   = (const int*)  d_in[2];
    const int*   positions = (const int*)  d_in[3];
    const float* emb       = (const float*)d_in[4];
    const float* t_w1      = (const float*)d_in[5];
    const float* t_b1      = (const float*)d_in[6];
    const float* t_w2      = (const float*)d_in[7];
    const float* t_b2      = (const float*)d_in[8];
    const float* ln1_w     = (const float*)d_in[9];
    const float* ln1_b     = (const float*)d_in[10];
    const float* qkv_w     = (const float*)d_in[11];
    const float* out_w     = (const float*)d_in[12];
    const float* ln2_w     = (const float*)d_in[13];
    const float* ln2_b     = (const float*)d_in[14];
    const float* mlp_w1    = (const float*)d_in[15];
    const float* mlp_b1    = (const float*)d_in[16];
    const float* mlp_w2    = (const float*)d_in[17];
    const float* mlp_b2    = (const float*)d_in[18];
    const float* ada_w     = (const float*)d_in[19];
    const float* ada_b     = (const float*)d_in[20];
    const float* lnf_w     = (const float*)d_in[21];
    const float* lnf_b     = (const float*)d_in[22];
    const float* fin_w     = (const float*)d_in[23];
    const float* fin_b     = (const float*)d_in[24];
    const float* fada_w    = (const float*)d_in[25];
    const float* fada_b    = (const float*)d_in[26];
    float* out = (float*)d_out;

    // --- workspace carve-out ---
    char* wp = (char*)d_ws;
    auto carve = [&](size_t bytes) -> char* {
        char* p = wp;
        wp += (bytes + 255) & ~(size_t)255;
        return p;
    };
    float*          x    = (float*)         carve((size_t)TT * DD * 4);        // residual stream
    float*          obuf = (float*)         carve((size_t)TT * DD * 4);        // attention output
    float*          big  = (float*)         carve((size_t)TT * 4 * DD * 4);    // qkv / mlp hidden
    unsigned short* abf  = (unsigned short*)carve((size_t)TT * 4 * DD * 2);    // bf16 activations
    unsigned short* wbf  = (unsigned short*)carve((size_t)VPAD * DD * 2);      // bf16 transposed weights
    float*          ada  = (float*)         carve((size_t)BB * 6 * DD * 4);    // adaLN (reused for fada)
    float*          cbuf = (float*)         carve((size_t)BB * CC * 4);        // conditioning
    float*          rsub = (float*)         carve((size_t)BB * 4);             // logit shift per seq

    const int nthr = 256;

    // embedding + conditioning
    embed_kernel<<<cdiv(TT * DD, nthr), nthr, 0, stream>>>(indices, emb, x);
    timestep_kernel<<<BB, CC, 0, stream>>>(sigma, t_w1, t_b1, t_w2, t_b2, cbuf, rsub);

    for (int l = 0; l < 2; ++l) {
        const float* qw  = qkv_w  + (size_t)l * DD * (3 * DD);
        const float* ow  = out_w  + (size_t)l * DD * DD;
        const float* m1w = mlp_w1 + (size_t)l * DD * (4 * DD);
        const float* m1b = mlp_b1 + (size_t)l * (4 * DD);
        const float* m2w = mlp_w2 + (size_t)l * (4 * DD) * DD;
        const float* m2b = mlp_b2 + (size_t)l * DD;
        const float* aw  = ada_w  + (size_t)l * CC * (6 * DD);
        const float* ab  = ada_b  + (size_t)l * (6 * DD);

        // adaLN params: [sh_msa | sc_msa | gt_msa | sh_mlp | sc_mlp | gt_mlp]
        small_mm_kernel<<<cdiv(BB * 6 * DD, nthr), nthr, 0, stream>>>(cbuf, aw, ab, ada, 6 * DD);

        // ---- attention branch ----
        ln_mod_kernel<<<TT, nthr, 0, stream>>>(x, ln1_w + l * DD, ln1_b + l * DD,
                                               ada, 6 * DD, 0, DD, seq_idx, abf);
        conv_tr_kernel<<<cdiv(3 * DD * DD, nthr), nthr, 0, stream>>>(qw, wbf, DD, 3 * DD, 3 * DD);
        gemm_bf16_wmma<0><<<dim3((3 * DD) / 128, TT / 64), nthr, 0, stream>>>(
            abf, wbf, nullptr, nullptr, nullptr, 0, 0, nullptr, nullptr, big, 3 * DD, DD);
        rope_kernel<<<cdiv(TT * 2 * HH * 32, nthr), nthr, 0, stream>>>(big, positions);
        attn_kernel<<<dim3(HH, BB), nthr, 0, stream>>>(big, obuf);
        f32_to_bf16_kernel<<<cdiv(TT * DD, nthr), nthr, 0, stream>>>(obuf, abf, TT * DD);
        conv_tr_kernel<<<cdiv(DD * DD, nthr), nthr, 0, stream>>>(ow, wbf, DD, DD, DD);
        // x = gt_msa * (o @ out_w) + x
        gemm_bf16_wmma<2><<<dim3(DD / 128, TT / 64), nthr, 0, stream>>>(
            abf, wbf, nullptr, seq_idx, ada, 2 * DD, 6 * DD, x, nullptr, x, DD, DD);

        // ---- MLP branch ----
        ln_mod_kernel<<<TT, nthr, 0, stream>>>(x, ln2_w + l * DD, ln2_b + l * DD,
                                               ada, 6 * DD, 3 * DD, 4 * DD, seq_idx, abf);
        conv_tr_kernel<<<cdiv(4 * DD * DD, nthr), nthr, 0, stream>>>(m1w, wbf, DD, 4 * DD, 4 * DD);
        gemm_bf16_wmma<1><<<dim3((4 * DD) / 128, TT / 64), nthr, 0, stream>>>(
            abf, wbf, m1b, nullptr, nullptr, 0, 0, nullptr, nullptr, big, 4 * DD, DD);
        f32_to_bf16_kernel<<<cdiv(TT * 4 * DD, nthr), nthr, 0, stream>>>(big, abf, TT * 4 * DD);
        conv_tr_kernel<<<cdiv(4 * DD * DD, nthr), nthr, 0, stream>>>(m2w, wbf, 4 * DD, DD, DD);
        // x = gt_mlp * (m @ mlp_w2 + b2) + x
        gemm_bf16_wmma<2><<<dim3(DD / 128, TT / 64), nthr, 0, stream>>>(
            abf, wbf, m2b, seq_idx, ada, 5 * DD, 6 * DD, x, nullptr, x, DD, 4 * DD);
    }

    // ---- final adaLN + logits ----
    small_mm_kernel<<<cdiv(BB * 2 * DD, nthr), nthr, 0, stream>>>(cbuf, fada_w, fada_b, ada, 2 * DD);
    ln_mod_kernel<<<TT, nthr, 0, stream>>>(x, lnf_w, lnf_b, ada, 2 * DD, 0, DD, seq_idx, abf);
    conv_tr_kernel<<<cdiv(VPAD * DD, nthr), nthr, 0, stream>>>(fin_w, wbf, DD, VV, VPAD);
    gemm_bf16_wmma<3><<<dim3(VPAD / 128, TT / 64), nthr, 0, stream>>>(
        abf, wbf, fin_b, seq_idx, nullptr, 0, 0, nullptr, rsub, out, VV, DD);
    set_targets_kernel<<<cdiv(TT, nthr), nthr, 0, stream>>>(out, indices);
}
